// ESMPNLayer_2774548873285
// MI455X (gfx1250) — matrix operand order repaired and checked by reading, not compiled, not run on hardware
//
#include <hip/hip_runtime.h>

#define HD 128
#define E_CNT 100000
#define ND0 50000
#define ND1 150000
#define ND2 50000

typedef __bf16 bf16;
typedef __attribute__((ext_vector_type(16))) __bf16 v16bf;
typedef __attribute__((ext_vector_type(8)))  float  v8f;

__device__ __forceinline__ float silu_f(float x)    { return x / (1.f + __expf(-x)); }
__device__ __forceinline__ float sigmoid_f(float x) { return 1.f / (1.f + __expf(-x)); }

__device__ __forceinline__ v8f vzero8() {
  v8f z = {0.f,0.f,0.f,0.f,0.f,0.f,0.f,0.f};
  return z;
}

// 16x32 bf16 A-fragment from an LDS row (row = lane%16).
// Element e -> k = kc*32 + (e/8)*16 + half*8 + (e%8): two contiguous 16B runs.
__device__ __forceinline__ v16bf ldsA(const bf16* rowbase, int kc, int half) {
  const uint4* p = reinterpret_cast<const uint4*>(rowbase + kc * 32 + half * 8);
  union { uint4 u[2]; v16bf v; } t;
  t.u[0] = p[0];
  t.u[1] = p[2];   // +16 halves = +32 bytes
  return t.v;
}

// Pre-packed B fragment: packed[((kc*8+nc)*32 + lane)*16 + e]
__device__ __forceinline__ v16bf ldB(const bf16* pack, int kc, int nc, int lane) {
  const v16bf* p = reinterpret_cast<const v16bf*>(pack + (((size_t)(kc * 8 + nc)) * 32 + lane) * 16);
  return *p;
}

#define WMMA_BF16(A, B, C) \
  __builtin_amdgcn_wmma_f32_16x16x32_bf16(false, (A), false, (B), (short)0, (C), false, false)

// ---------------------------------------------------------------------------
// Repack fp32 W[KR][128] into bf16 B-fragment layout, K padded to KC*32.
// B element (lane,e) of tile (kc,nc): k = kc*32 + (e/8)*16 + (lane/16)*8 + (e%8), n = nc*16 + lane%16.
// ---------------------------------------------------------------------------
__global__ void repack_w(const float* __restrict__ W, bf16* __restrict__ P,
                         int KR, int KC) {
  int t = blockIdx.x * blockDim.x + threadIdx.x;
  int total = KC * 4096;
  if (t >= total) return;
  int e    = t & 15;
  int lane = (t >> 4) & 31;
  int nc   = (t >> 9) & 7;
  int kc   = t >> 12;
  int k = kc * 32 + ((e >> 3) << 4) + ((lane >> 4) << 3) + (e & 7);
  int n = nc * 16 + (lane & 15);
  float v = (k < KR) ? W[(size_t)k * HD + n] : 0.f;
  P[t] = (bf16)v;
}

__global__ void zero_f(float* __restrict__ p, long long n) {
  long long t = (long long)blockIdx.x * blockDim.x + threadIdx.x;
  if (t < n) p[t] = 0.f;
}

// ---------------------------------------------------------------------------
// Edge message kernel: one wave32 per TWO 16-edge tiles (B fragments reused
// across both tiles to halve weight traffic).
// ---------------------------------------------------------------------------
__global__ __launch_bounds__(32)
void edge_msg(const float* __restrict__ xs_s, const float* __restrict__ xs_r,
              const int* __restrict__ adj, const float* __restrict__ inv, int kinv,
              const bf16* __restrict__ w1p, const float* __restrict__ b1,
              const bf16* __restrict__ w2p, const float* __restrict__ b2,
              const float* __restrict__ ew_w, const float* __restrict__ ew_b,
              float* __restrict__ agg, float* __restrict__ mes_out) {
  __shared__ __align__(16) bf16 sState[2][16 * 288];
  __shared__ __align__(16) bf16 sHid[2][16 * 128];
  __shared__ int sIdx[64];
  const int lane = threadIdx.x;
  const int e00 = blockIdx.x * 32;

  {
    int e = min(e00 + lane, E_CNT - 1);
    sIdx[lane]      = adj[e];          // send
    sIdx[32 + lane] = adj[E_CNT + e];  // rec
  }
  __syncthreads();

#pragma unroll
  for (int t = 0; t < 2; ++t) {
    for (int r = 0; r < 16; ++r) {
      int rr = t * 16 + r;
      int e = min(e00 + rr, E_CNT - 1);
      int s = sIdx[rr], rc = sIdx[32 + rr];
      const float4* ps = reinterpret_cast<const float4*>(xs_s + (long long)s * HD);
      const float4* pr = reinterpret_cast<const float4*>(xs_r + (long long)rc * HD);
      float4 a4 = ps[lane];
      float4 b4 = pr[lane];
      int c = lane * 4;
      sState[t][r * 288 + c + 0] = (bf16)a4.x;  sState[t][r * 288 + c + 1] = (bf16)a4.y;
      sState[t][r * 288 + c + 2] = (bf16)a4.z;  sState[t][r * 288 + c + 3] = (bf16)a4.w;
      sState[t][r * 288 + 128 + c + 0] = (bf16)b4.x;  sState[t][r * 288 + 128 + c + 1] = (bf16)b4.y;
      sState[t][r * 288 + 128 + c + 2] = (bf16)b4.z;  sState[t][r * 288 + 128 + c + 3] = (bf16)b4.w;
      float iv = (lane < kinv) ? inv[(long long)e * kinv + lane] : 0.f;
      sState[t][r * 288 + 256 + lane] = (bf16)iv;
    }
  }
  __syncthreads();

  const int half = lane >> 4;
  const int nlo  = lane & 15;

  // layer 1: K=288 (9 chunks), N=128 (8 tiles), 2 M-tiles share each B frag
  v8f acc[2][8];
#pragma unroll
  for (int t = 0; t < 2; ++t)
#pragma unroll
    for (int nc = 0; nc < 8; ++nc) acc[t][nc] = vzero8();
  const bf16* rowA0 = &sState[0][nlo * 288];
  const bf16* rowA1 = &sState[1][nlo * 288];
  for (int kc = 0; kc < 9; ++kc) {
    v16bf a0 = ldsA(rowA0, kc, half);
    v16bf a1 = ldsA(rowA1, kc, half);
#pragma unroll
    for (int nc = 0; nc < 8; ++nc) {
      v16bf b = ldB(w1p, kc, nc, lane);
      acc[0][nc] = WMMA_BF16(a0, b, acc[0][nc]);
      acc[1][nc] = WMMA_BF16(a1, b, acc[1][nc]);
    }
  }
#pragma unroll
  for (int t = 0; t < 2; ++t)
#pragma unroll
    for (int nc = 0; nc < 8; ++nc) {
      int n = nc * 16 + nlo;
      float bias = b1[n];
#pragma unroll
      for (int r = 0; r < 8; ++r)
        sHid[t][(r + half * 8) * 128 + n] = (bf16)silu_f(acc[t][nc][r] + bias);
    }
  __syncthreads();

  // layer 2: K=128 (4 chunks)
  v8f acc2[2][8];
#pragma unroll
  for (int t = 0; t < 2; ++t)
#pragma unroll
    for (int nc = 0; nc < 8; ++nc) acc2[t][nc] = vzero8();
  const bf16* rowH0 = &sHid[0][nlo * 128];
  const bf16* rowH1 = &sHid[1][nlo * 128];
  for (int kc = 0; kc < 4; ++kc) {
    v16bf a0 = ldsA(rowH0, kc, half);
    v16bf a1 = ldsA(rowH1, kc, half);
#pragma unroll
    for (int nc = 0; nc < 8; ++nc) {
      v16bf b = ldB(w2p, kc, nc, lane);
      acc2[0][nc] = WMMA_BF16(a0, b, acc2[0][nc]);
      acc2[1][nc] = WMMA_BF16(a1, b, acc2[1][nc]);
    }
  }

  // bias + outer silu (in place), edge-gate dot products
  float part[2][8];
#pragma unroll
  for (int t = 0; t < 2; ++t)
#pragma unroll
    for (int r = 0; r < 8; ++r) part[t][r] = 0.f;
#pragma unroll
  for (int nc = 0; nc < 8; ++nc) {
    int n = nc * 16 + nlo;
    float bias = b2[n];
    float w = ew_w[n];
#pragma unroll
    for (int t = 0; t < 2; ++t)
#pragma unroll
      for (int r = 0; r < 8; ++r) {
        float v = silu_f(acc2[t][nc][r] + bias);
        acc2[t][nc][r] = v;
        part[t][r] += v * w;
      }
  }
  float ebv = ew_b[0];
#pragma unroll
  for (int t = 0; t < 2; ++t)
#pragma unroll
    for (int r = 0; r < 8; ++r) {
      float p = part[t][r];
      p += __shfl_xor(p, 1, 32);
      p += __shfl_xor(p, 2, 32);
      p += __shfl_xor(p, 4, 32);
      p += __shfl_xor(p, 8, 32);
      part[t][r] = sigmoid_f(p + ebv);
    }

  // scatter: agg[rec] += m * ew ; store mes for 0_0
#pragma unroll
  for (int t = 0; t < 2; ++t)
#pragma unroll
    for (int r = 0; r < 8; ++r) {
      int m = r + half * 8;
      int rr = t * 16 + m;
      int e = e00 + rr;
      if (e >= E_CNT) continue;
      int rc = sIdx[32 + rr];
      float w = part[t][r];
      float* drow = agg + (long long)rc * HD;
#pragma unroll
      for (int nc = 0; nc < 8; ++nc) {
        int n = nc * 16 + nlo;
        float v = acc2[t][nc][r];
        atomicAdd(drow + n, v * w);
        if (mes_out) mes_out[(long long)e * HD + n] = v;
      }
    }
}

// ---------------------------------------------------------------------------
// Node update: h = [x | agg...] (K = KP), xd = x + MLP2(h), out = layernorm(xd)
// Two 16-node tiles per wave.
// ---------------------------------------------------------------------------
__global__ __launch_bounds__(32)
void node_update(const float* __restrict__ xd,
                 const float* __restrict__ agg0, const float* __restrict__ agg1,
                 const float* __restrict__ agg2, int KP,
                 const bf16* __restrict__ w1p, const float* __restrict__ b1,
                 const bf16* __restrict__ w2p, const float* __restrict__ b2,
                 const float* __restrict__ g, const float* __restrict__ bb,
                 float* __restrict__ out, int Nn) {
  __shared__ __align__(16) bf16 sIn[2][16 * 512];
  __shared__ __align__(16) bf16 sHid[2][16 * 128];
  __shared__ __align__(16) float sM[2][16 * 128];
  const int lane = threadIdx.x;
  const int n0 = blockIdx.x * 32;
  const int nAgg = KP / 128 - 1;
  const float* srcs[3] = {agg0, agg1, agg2};

#pragma unroll
  for (int t = 0; t < 2; ++t) {
    for (int r = 0; r < 16; ++r) {
      int node = min(n0 + t * 16 + r, Nn - 1);
      {
        const float4* p = reinterpret_cast<const float4*>(xd + (long long)node * HD);
        float4 v = p[lane];
        int c = lane * 4;
        sIn[t][r * KP + c + 0] = (bf16)v.x;  sIn[t][r * KP + c + 1] = (bf16)v.y;
        sIn[t][r * KP + c + 2] = (bf16)v.z;  sIn[t][r * KP + c + 3] = (bf16)v.w;
      }
      for (int s = 0; s < nAgg; ++s) {
        const float4* p = reinterpret_cast<const float4*>(srcs[s] + (long long)node * HD);
        float4 v = p[lane];
        int c = (s + 1) * 128 + lane * 4;
        sIn[t][r * KP + c + 0] = (bf16)v.x;  sIn[t][r * KP + c + 1] = (bf16)v.y;
        sIn[t][r * KP + c + 2] = (bf16)v.z;  sIn[t][r * KP + c + 3] = (bf16)v.w;
      }
    }
  }
  __syncthreads();

  const int half = lane >> 4;
  const int nlo  = lane & 15;
  const int KC   = KP / 32;

  v8f acc[2][8];
#pragma unroll
  for (int t = 0; t < 2; ++t)
#pragma unroll
    for (int nc = 0; nc < 8; ++nc) acc[t][nc] = vzero8();
  const bf16* rowA0 = &sIn[0][nlo * KP];
  const bf16* rowA1 = &sIn[1][nlo * KP];
  for (int kc = 0; kc < KC; ++kc) {
    v16bf a0 = ldsA(rowA0, kc, half);
    v16bf a1 = ldsA(rowA1, kc, half);
#pragma unroll
    for (int nc = 0; nc < 8; ++nc) {
      v16bf b = ldB(w1p, kc, nc, lane);
      acc[0][nc] = WMMA_BF16(a0, b, acc[0][nc]);
      acc[1][nc] = WMMA_BF16(a1, b, acc[1][nc]);
    }
  }
#pragma unroll
  for (int t = 0; t < 2; ++t)
#pragma unroll
    for (int nc = 0; nc < 8; ++nc) {
      int n = nc * 16 + nlo;
      float bias = b1[n];
#pragma unroll
      for (int r = 0; r < 8; ++r)
        sHid[t][(r + half * 8) * 128 + n] = (bf16)silu_f(acc[t][nc][r] + bias);
    }
  __syncthreads();

  v8f acc2[2][8];
#pragma unroll
  for (int t = 0; t < 2; ++t)
#pragma unroll
    for (int nc = 0; nc < 8; ++nc) acc2[t][nc] = vzero8();
  const bf16* rowH0 = &sHid[0][nlo * 128];
  const bf16* rowH1 = &sHid[1][nlo * 128];
  for (int kc = 0; kc < 4; ++kc) {
    v16bf a0 = ldsA(rowH0, kc, half);
    v16bf a1 = ldsA(rowH1, kc, half);
#pragma unroll
    for (int nc = 0; nc < 8; ++nc) {
      v16bf b = ldB(w2p, kc, nc, lane);
      acc2[0][nc] = WMMA_BF16(a0, b, acc2[0][nc]);
      acc2[1][nc] = WMMA_BF16(a1, b, acc2[1][nc]);
    }
  }
#pragma unroll
  for (int t = 0; t < 2; ++t)
#pragma unroll
    for (int nc = 0; nc < 8; ++nc) {
      int n = nc * 16 + nlo;
      float bias = b2[n];
#pragma unroll
      for (int r = 0; r < 8; ++r)
        sM[t][(r + half * 8) * 128 + n] = acc2[t][nc][r] + bias;
    }
  __syncthreads();

  // residual + layernorm: 2 lanes per row (each handles 64 cols)
#pragma unroll
  for (int t = 0; t < 2; ++t) {
    int row = nlo;
    int node = n0 + t * 16 + row;
    int cn = min(node, Nn - 1);
    const float* xr = xd + (long long)cn * HD;
    float s = 0.f, ss = 0.f;
    for (int j = 0; j < 64; ++j) {
      int c = half * 64 + j;
      float v = xr[c] + sM[t][row * 128 + c];
      s += v;  ss += v * v;
    }
    s  += __shfl_xor(s, 16, 32);
    ss += __shfl_xor(ss, 16, 32);
    float mean = s * (1.f / HD);
    float var  = ss * (1.f / HD) - mean * mean;
    float rstd = rsqrtf(var + 1e-5f);
    if (node < Nn) {
      float* orow = out + (long long)node * HD;
      for (int j = 0; j < 64; ++j) {
        int c = half * 64 + j;
        float v = xr[c] + sM[t][row * 128 + c];
        orow[c] = (v - mean) * rstd * g[c] + bb[c];
      }
    }
  }
}

// ---------------------------------------------------------------------------
// Position head over 0_0 edges: pm = MLP2(mes00); accLoc[rec] += (pos[s]-pos[r])*pm
// ---------------------------------------------------------------------------
__global__ __launch_bounds__(32)
void pos_edge(const float* __restrict__ mes00, const int* __restrict__ adj,
              const bf16* __restrict__ w1p, const float* __restrict__ b1,
              const float* __restrict__ w2, const float* __restrict__ b2,
              const float* __restrict__ pos0, float* __restrict__ accLoc) {
  __shared__ __align__(16) bf16 sIn[16 * 128];
  __shared__ __align__(16) bf16 sHid[16 * 128];
  const int lane = threadIdx.x;
  const int e0 = blockIdx.x * 16;

  for (int r = 0; r < 16; ++r) {
    int e = min(e0 + r, E_CNT - 1);
    const float4* p = reinterpret_cast<const float4*>(mes00 + (long long)e * HD);
    float4 v = p[lane];
    int c = lane * 4;
    sIn[r * 128 + c + 0] = (bf16)v.x;  sIn[r * 128 + c + 1] = (bf16)v.y;
    sIn[r * 128 + c + 2] = (bf16)v.z;  sIn[r * 128 + c + 3] = (bf16)v.w;
  }
  __syncthreads();

  const int half = lane >> 4;
  const int nlo  = lane & 15;
  v8f acc[8];
#pragma unroll
  for (int nc = 0; nc < 8; ++nc) acc[nc] = vzero8();
  const bf16* rowA = sIn + nlo * 128;
  for (int kc = 0; kc < 4; ++kc) {
    v16bf a = ldsA(rowA, kc, half);
#pragma unroll
    for (int nc = 0; nc < 8; ++nc) {
      v16bf b = ldB(w1p, kc, nc, lane);
      acc[nc] = WMMA_BF16(a, b, acc[nc]);
    }
  }
#pragma unroll
  for (int nc = 0; nc < 8; ++nc) {
    int n = nc * 16 + nlo;
    float bias = b1[n];
#pragma unroll
    for (int r = 0; r < 8; ++r)
      sHid[(r + half * 8) * 128 + n] = (bf16)silu_f(acc[nc][r] + bias);
  }
  __syncthreads();

  float p = 0.f;
  for (int j = 0; j < 64; ++j) {
    int c = half * 64 + j;
    p += (float)sHid[nlo * 128 + c] * w2[c];
  }
  p += __shfl_xor(p, 16, 32);
  float pm = p + b2[0];

  if (lane < 16) {
    int e = e0 + lane;
    if (e < E_CNT) {
      int s = adj[e], rc = adj[E_CNT + e];
#pragma unroll
      for (int c = 0; c < 3; ++c) {
        float d = pos0[(long long)s * 3 + c] - pos0[(long long)rc * 3 + c];
        atomicAdd(accLoc + (long long)rc * 3 + c, d * pm);
      }
    }
  }
}

// ---------------------------------------------------------------------------
// Velocity head: vm = MLP2(out0); loc = pos0 + accLoc + vm*vel0
// ---------------------------------------------------------------------------
__global__ __launch_bounds__(32)
void vel_node(const float* __restrict__ out0,
              const bf16* __restrict__ w1p, const float* __restrict__ b1,
              const float* __restrict__ w2, const float* __restrict__ b2,
              const float* __restrict__ pos0, const float* __restrict__ vel0,
              const float* __restrict__ accLoc, float* __restrict__ loc) {
  __shared__ __align__(16) bf16 sIn[16 * 128];
  __shared__ __align__(16) bf16 sHid[16 * 128];
  const int lane = threadIdx.x;
  const int n0 = blockIdx.x * 16;

  for (int r = 0; r < 16; ++r) {
    int node = min(n0 + r, ND0 - 1);
    const float4* p = reinterpret_cast<const float4*>(out0 + (long long)node * HD);
    float4 v = p[lane];
    int c = lane * 4;
    sIn[r * 128 + c + 0] = (bf16)v.x;  sIn[r * 128 + c + 1] = (bf16)v.y;
    sIn[r * 128 + c + 2] = (bf16)v.z;  sIn[r * 128 + c + 3] = (bf16)v.w;
  }
  __syncthreads();

  const int half = lane >> 4;
  const int nlo  = lane & 15;
  v8f acc[8];
#pragma unroll
  for (int nc = 0; nc < 8; ++nc) acc[nc] = vzero8();
  const bf16* rowA = sIn + nlo * 128;
  for (int kc = 0; kc < 4; ++kc) {
    v16bf a = ldsA(rowA, kc, half);
#pragma unroll
    for (int nc = 0; nc < 8; ++nc) {
      v16bf b = ldB(w1p, kc, nc, lane);
      acc[nc] = WMMA_BF16(a, b, acc[nc]);
    }
  }
#pragma unroll
  for (int nc = 0; nc < 8; ++nc) {
    int n = nc * 16 + nlo;
    float bias = b1[n];
#pragma unroll
    for (int r = 0; r < 8; ++r)
      sHid[(r + half * 8) * 128 + n] = (bf16)silu_f(acc[nc][r] + bias);
  }
  __syncthreads();

  float p = 0.f;
  for (int j = 0; j < 64; ++j) {
    int c = half * 64 + j;
    p += (float)sHid[nlo * 128 + c] * w2[c];
  }
  p += __shfl_xor(p, 16, 32);
  float vm = p + b2[0];

  if (lane < 16) {
    int node = n0 + lane;
    if (node < ND0) {
#pragma unroll
      for (int c = 0; c < 3; ++c) {
        long long i = (long long)node * 3 + c;
        loc[i] = pos0[i] + accLoc[i] + vm * vel0[i];
      }
    }
  }
}

// ---------------------------------------------------------------------------
extern "C" void kernel_launch(void* const* d_in, const int* in_sizes, int n_in,
                              void* d_out, int out_size, void* d_ws, size_t ws_size,
                              hipStream_t stream) {
  (void)n_in; (void)out_size; (void)ws_size;
  static const int INVC[6] = {3, 3, 3, 6, 6, 6};

  const float *x0, *x1, *x2, *pos0, *vel0;
  const int* adj[6];
  const float* inv[6];
  const float *mw1[6], *mb1[6], *mw2[6], *mb2[6], *ew[6], *eb[6];
  const float *uw1[3], *ub1[3], *uw2[3], *ub2[3], *ng[3], *nb[3];
  const float *pw1, *pb1, *pw2, *pb2, *vw1, *vb1, *vw2, *vb2;

  auto F = [&](int i) -> const float* { return (const float*)d_in[i]; };
  auto I = [&](int i) -> const int*   { return (const int*)d_in[i]; };

  auto sortedParams = [&](int pb) {
    for (int a = 0; a < 6; ++a) {
      int b0 = pb + a * 6;
      eb[a] = F(b0); ew[a] = F(b0 + 1); mb1[a] = F(b0 + 2);
      mb2[a] = F(b0 + 3); mw1[a] = F(b0 + 4); mw2[a] = F(b0 + 5);
    }
    int nbase = pb + 36;
    for (int d = 0; d < 3; ++d) { nb[d] = F(nbase + d * 2); ng[d] = F(nbase + d * 2 + 1); }
    int pbase = nbase + 6;
    pb1 = F(pbase); pb2 = F(pbase + 1); pw1 = F(pbase + 2); pw2 = F(pbase + 3);
    vb1 = F(pbase + 4); vb2 = F(pbase + 5); vw1 = F(pbase + 6); vw2 = F(pbase + 7);
    int ubase = pbase + 8;
    for (int d = 0; d < 3; ++d) {
      ub1[d] = F(ubase + d * 4); ub2[d] = F(ubase + d * 4 + 1);
      uw1[d] = F(ubase + d * 4 + 2); uw2[d] = F(ubase + d * 4 + 3);
    }
    return ubase + 12;
  };
  auto insertionParams = [&](int pb) {
    for (int a = 0; a < 6; ++a) {
      int b0 = pb + a * 6;
      mw1[a] = F(b0); mb1[a] = F(b0 + 1); mw2[a] = F(b0 + 2);
      mb2[a] = F(b0 + 3); ew[a] = F(b0 + 4); eb[a] = F(b0 + 5);
    }
    int ubase = pb + 36;
    for (int d = 0; d < 3; ++d) {
      uw1[d] = F(ubase + d * 4); ub1[d] = F(ubase + d * 4 + 1);
      uw2[d] = F(ubase + d * 4 + 2); ub2[d] = F(ubase + d * 4 + 3);
    }
    int nbase = ubase + 12;
    for (int d = 0; d < 3; ++d) { ng[d] = F(nbase + d * 2); nb[d] = F(nbase + d * 2 + 1); }
    int pbase = nbase + 6;
    pw1 = F(pbase); pb1 = F(pbase + 1); pw2 = F(pbase + 2); pb2 = F(pbase + 3);
    vw1 = F(pbase + 4); vb1 = F(pbase + 5); vw2 = F(pbase + 6); vb2 = F(pbase + 7);
    return pbase + 8;
  };

  if (in_sizes[0] == 2 * E_CNT) {
    // fully tree-sorted: adj*, inv*, params(sorted), pos0, vel0, x0, x1, x2
    for (int a = 0; a < 6; ++a) { adj[a] = I(a); inv[a] = F(6 + a); }
    int xbase = sortedParams(12);
    pos0 = F(xbase); vel0 = F(xbase + 1);
    x0 = F(xbase + 2); x1 = F(xbase + 3); x2 = F(xbase + 4);
  } else {
    x0 = F(0); x1 = F(1); x2 = F(2); pos0 = F(3); vel0 = F(4);
    if (in_sizes[6] == 3 * E_CNT) {     // interleaved adj,inv per adjacency
      for (int a = 0; a < 6; ++a) { adj[a] = I(5 + 2 * a); inv[a] = F(6 + 2 * a); }
    } else {                            // grouped: all adj then all inv
      for (int a = 0; a < 6; ++a) { adj[a] = I(5 + a); inv[a] = F(11 + a); }
    }
    if (in_sizes[17] == 1) sortedParams(17);
    else                   insertionParams(17);
  }

  // ---- workspace carve (bytes) ----
  char* ws = (char*)d_ws;
  float* g00 = (float*)(ws + 0LL);            // N0*128
  float* g10 = (float*)(ws + 25600000LL);     // N0*128
  float* g01 = (float*)(ws + 51200000LL);     // N1*128
  float* g11 = (float*)(ws + 128000000LL);    // N1*128
  float* g21 = (float*)(ws + 204800000LL);    // N1*128
  float* g12 = (float*)(ws + 281600000LL);    // N2*128
  float* mes = (float*)(ws + 307200000LL);    // E*128
  float* acc = (float*)(ws + 358400000LL);    // N0*3
  char* wp = ws + 359000000LL;
  bf16 *mw1p[6], *mw2p[6], *uw1p[3], *uw2p[3], *pw1p, *vw1p;
  size_t o = 0;
  for (int a = 0; a < 6; ++a) {
    mw1p[a] = (bf16*)(wp + o); o += 73728;   // 9 kc
    mw2p[a] = (bf16*)(wp + o); o += 32768;   // 4 kc
  }
  static const size_t UW1SZ[3] = {98304, 131072, 65536};
  for (int d = 0; d < 3; ++d) {
    uw1p[d] = (bf16*)(wp + o); o += UW1SZ[d];
    uw2p[d] = (bf16*)(wp + o); o += 32768;
  }
  pw1p = (bf16*)(wp + o); o += 32768;
  vw1p = (bf16*)(wp + o); o += 32768;

  // ---- 1. repack weights to bf16 fragment layout ----
  for (int a = 0; a < 6; ++a) {
    repack_w<<<(9 * 4096 + 255) / 256, 256, 0, stream>>>(mw1[a], mw1p[a], 256 + INVC[a], 9);
    repack_w<<<(4 * 4096 + 255) / 256, 256, 0, stream>>>(mw2[a], mw2p[a], 128, 4);
  }
  static const int FH[3]  = {384, 512, 256};
  static const int KCU[3] = {12, 16, 8};
  for (int d = 0; d < 3; ++d) {
    repack_w<<<(KCU[d] * 4096 + 255) / 256, 256, 0, stream>>>(uw1[d], uw1p[d], FH[d], KCU[d]);
    repack_w<<<(4 * 4096 + 255) / 256, 256, 0, stream>>>(uw2[d], uw2p[d], 128, 4);
  }
  repack_w<<<64, 256, 0, stream>>>(pw1, pw1p, 128, 4);
  repack_w<<<64, 256, 0, stream>>>(vw1, vw1p, 128, 4);

  // ---- 2. zero aggregation buffers + loc accumulator ----
  long long aggN = 307200000LL / 4;
  zero_f<<<(int)((aggN + 255) / 256), 256, 0, stream>>>(g00, aggN);
  zero_f<<<(ND0 * 3 + 255) / 256, 256, 0, stream>>>(acc, (long long)ND0 * 3);

  // ---- 3. edge message kernels (2 tiles per wave) ----
  const float* XS[3] = {x0, x1, x2};
  struct { int s, r; float* agg; } AD[6] = {
      {0, 0, g00}, {0, 1, g01}, {1, 0, g10}, {1, 1, g11}, {1, 2, g12}, {2, 1, g21}};
  int eblocks = (E_CNT + 31) / 32;
  for (int a = 0; a < 6; ++a) {
    edge_msg<<<eblocks, 32, 0, stream>>>(
        XS[AD[a].s], XS[AD[a].r], adj[a], inv[a], INVC[a],
        mw1p[a], mb1[a], mw2p[a], mb2[a], ew[a], eb[a],
        AD[a].agg, (a == 0) ? mes : nullptr);
  }

  // ---- 4. node updates (2 tiles per wave) ----
  float* out  = (float*)d_out;
  float* out0 = out;
  float* out1 = out + 6400000LL;
  float* out2 = out + 25600000LL;
  float* loc  = out + 32000000LL;
  node_update<<<(ND0 + 31) / 32, 32, 0, stream>>>(x0, g00, g10, nullptr, 384,
      uw1p[0], ub1[0], uw2p[0], ub2[0], ng[0], nb[0], out0, ND0);
  node_update<<<(ND1 + 31) / 32, 32, 0, stream>>>(x1, g01, g11, g21, 512,
      uw1p[1], ub1[1], uw2p[1], ub2[1], ng[1], nb[1], out1, ND1);
  node_update<<<(ND2 + 31) / 32, 32, 0, stream>>>(x2, g12, nullptr, nullptr, 256,
      uw1p[2], ub1[2], uw2p[2], ub2[2], ng[2], nb[2], out2, ND2);

  // ---- 5. equivariant position update ----
  pos_edge<<<(E_CNT + 15) / 16, 32, 0, stream>>>(mes, adj[0], pw1p, pb1, pw2, pb2, pos0, acc);
  vel_node<<<(ND0 + 15) / 16, 32, 0, stream>>>(out0, vw1p, vb1, vw2, vb2,
                                               pos0, vel0, acc, loc);
}
